// DLRM_35313221108343
// MI455X (gfx1250) — matrix operand order repaired and checked by reading, not compiled, not run on hardware
//
#include <hip/hip_runtime.h>
#include <hip/hip_bf16.h>

#define Dm 28
#define Lh 50
#define CDm 196
#define PAD_IDX 59047
#define UPAD_IDX 162541

typedef _Float16 f16;
typedef __attribute__((ext_vector_type(16))) _Float16 v16h;
typedef __attribute__((ext_vector_type(8)))  _Float16 v8h;
typedef __attribute__((ext_vector_type(4)))  _Float16 v4h;
typedef __attribute__((ext_vector_type(8)))  float    v8f;

#define WMMA_F16(a, b, c) \
  __builtin_amdgcn_wmma_f32_16x16x32_f16(false, (a), false, (b), (short)0, (c), false, false)

#define LDSTR 40  // LDS row stride in halves (80 B, keeps 16-B alignment for b128 loads)

// ---------------- WMMA fragment loaders (ISA 7.12.2 layouts, wave32) ----------------
// A 16x32 f16: lanes 0-15 hold row M=lane, K 0-7 & 16-23; lanes 16-31 hold K 8-15 & 24-31.
// Row-major LDS tile -> each lane reads two contiguous 8-half runs (2x ds_load_b128).
__device__ __forceinline__ v16h frag_a(const f16* As, int rowbase) {
  int lane = threadIdx.x & 31;
  int hi = lane >> 4;
  const f16* p = As + (rowbase + (lane & 15)) * LDSTR;
  v8h lo = *(const v8h*)(p + hi * 8);
  v8h hh = *(const v8h*)(p + 16 + hi * 8);
  return __builtin_shufflevector(lo, hh, 0, 1, 2, 3, 4, 5, 6, 7, 8, 9, 10, 11, 12, 13, 14, 15);
}
// B 32x16 f16: lanes 0-15 hold col N=lane, K 0-7 & 16-23; lanes 16-31 hold K 8-15 & 24-31.
// LDS tile stored TRANSPOSED (BsT[n][k]) -> per-lane two contiguous 8-half runs (2x ds_load_b128).
__device__ __forceinline__ v16h frag_bT(const f16* BsT, int colbase) {
  int lane = threadIdx.x & 31;
  const f16* p = BsT + (colbase + (lane & 15)) * LDSTR + (lane >> 4) * 8;
  v8h lo = *(const v8h*)(p);
  v8h hh = *(const v8h*)(p + 16);
  return __builtin_shufflevector(lo, hh, 0, 1, 2, 3, 4, 5, 6, 7, 8, 9, 10, 11, 12, 13, 14, 15);
}

__device__ __forceinline__ float act_f(float v, int act) {
  if (act == 1) return v > 0.f ? v : 0.f;
  if (act == 2) return __builtin_amdgcn_rcpf(1.f + __expf(-v));  // fast sigmoid (v_rcp_f32)
  return v;
}

// ---------------- Generic GEMM: C[M,N] = act(A[M,K] @ W[K,N] + bias) ----------------
// f32 in/out, f16 WMMA compute. 256 threads = 8 waves; tile 32(M) x 64(N), K step 32.
// Staging: clamped-address loads (no divergence), vector LDS writes, one shot per k-step.
__global__ void gemm_bias_act(const float* __restrict__ A, int lda,
                              const float* __restrict__ Bw, int ldb,
                              const float* __restrict__ bias,
                              float* __restrict__ C, int ldc,
                              int M, int N, int K, int act) {
  __shared__ f16 As[32 * LDSTR];
  __shared__ f16 BsT[64 * LDSTR];
  const int tid = threadIdx.x;
  const int wave = tid >> 5;
  const int wm = wave & 1, wn = wave >> 1;
  const int m0 = blockIdx.x * 32;
  const int n0 = blockIdx.y * 64;
  // A staging map: row = tid>>3 (0..31), 4 consecutive k per thread
  const int ar = tid >> 3;
  const int ac0 = (tid & 7) * 4;
  const int am = m0 + ar;
  const size_t amc = (size_t)(am < M ? am : M - 1) * lda;
  // B staging map: n = tid&63, 8 consecutive k per thread
  const int bn = tid & 63;
  const int bk0 = (tid >> 6) * 8;
  const int bnc = (n0 + bn) < N ? (n0 + bn) : N - 1;
  const bool bnv = (n0 + bn) < N;

  v8f c = {0.f, 0.f, 0.f, 0.f, 0.f, 0.f, 0.f, 0.f};
  for (int k0 = 0; k0 < K; k0 += 32) {
    float av[4];
#pragma unroll
    for (int t = 0; t < 4; ++t) {
      int k = k0 + ac0 + t;
      av[t] = A[amc + (k < K ? k : K - 1)];
      if (am >= M || k >= K) av[t] = 0.f;
    }
    float bv[8];
#pragma unroll
    for (int t = 0; t < 8; ++t) {
      int k = k0 + bk0 + t;
      bv[t] = Bw[(size_t)(k < K ? k : K - 1) * ldb + bnc];
      if (k >= K || !bnv) bv[t] = 0.f;
    }
    __syncthreads();  // previous iteration's fragment reads complete
    v4h pa;
#pragma unroll
    for (int t = 0; t < 4; ++t) pa[t] = (f16)av[t];
    *(v4h*)(As + ar * LDSTR + ac0) = pa;
    v8h pb;
#pragma unroll
    for (int t = 0; t < 8; ++t) pb[t] = (f16)bv[t];
    *(v8h*)(BsT + bn * LDSTR + bk0) = pb;
    __syncthreads();
    v16h a = frag_a(As, wm * 16);
    v16h b = frag_bT(BsT, wn * 16);
    c = WMMA_F16(a, b, c);
  }
  const int lane = tid & 31;
  const int rbase = (lane >> 4) * 8;
  const int n = n0 + wn * 16 + (lane & 15);
  // hoisted, branchless bias load (invariant over r)
  float bvn = 0.f;
  if (bias) bvn = bias[n < N ? n : N - 1];
#pragma unroll
  for (int r = 0; r < 8; ++r) {
    int m = m0 + wm * 16 + rbase + r;
    if (m < M && n < N) {
      C[(size_t)m * ldc + n] = act_f(c[r] + bvn, act);
    }
  }
}

// ---------------- Fused DIN logits ----------------
// logits[m] = relu([s, s*t] @ W1' + t@W1_t + b1) @ w2 + b2, A built on the fly (branchless).
// s = [seqv[b,l,:], ratings[b,l]*rw + rb], t = [tvec[b,:], tvec[b,:]] folded into tbp.
__global__ void din_logits(const float* __restrict__ seqv,   // [B,L,D]
                           const float* __restrict__ rat,    // [B,L]
                           const float* __restrict__ rw, const float* __restrict__ rb,
                           const float* __restrict__ tvec,   // [B,D]
                           const float* __restrict__ w1,     // [168,64]
                           const float* __restrict__ b1,     // [64]
                           const float* __restrict__ w2,     // [64]
                           const float* __restrict__ b2,     // [1]
                           const float* __restrict__ tbp,    // [B,64]  (t @ W1[56:112])
                           float* __restrict__ logits, int Mrows) {
  __shared__ f16 As[16 * LDSTR];
  __shared__ f16 BsT[64 * LDSTR];
  __shared__ float hid[16][64];
  const int tid = threadIdx.x;
  const int wave = tid >> 5;  // 0..3 -> N tile
  const int m0 = blockIdx.x * 16;
  // A staging map: row = tid>>3 (0..15), 4 consecutive k per thread
  const int ar = tid >> 3;
  const int ac0 = (tid & 7) * 4;
  const int am = m0 + ar;
  const int amc = am < Mrows ? am : Mrows - 1;
  const int ab = amc / Lh;
  const size_t abl = (size_t)amc;  // == b*Lh + l
  // B staging map: n = tid&63, 16 consecutive k per thread
  const int bn = tid & 63;
  const int bk0 = (tid >> 6) * 16;  // 0 or 16

  const float ratv = rat[abl];
  v8f c = {0.f, 0.f, 0.f, 0.f, 0.f, 0.f, 0.f, 0.f};
  for (int k0 = 0; k0 < 128; k0 += 32) {
    v4h pa;
#pragma unroll
    for (int t = 0; t < 4; ++t) {
      int k = k0 + ac0 + t;
      int kc = k < 112 ? k : 0;
      int reg = kc / 28;
      int j = kc - reg * 28;
      float sv = seqv[abl * Dm + j];
      float rv = ratv * rw[j] + rb[j];
      float tv = tvec[(size_t)ab * Dm + j];
      float base = (reg & 1) ? rv : sv;
      float v = (reg & 2) ? base * tv : base;
      if (k >= 112 || am >= Mrows) v = 0.f;
      pa[t] = (f16)v;
    }
    f16 bvals[16];
#pragma unroll
    for (int q = 0; q < 16; ++q) {
      int k = k0 + bk0 + q;
      int krow = (k < 56) ? k : k + 56;   // skip t-slice rows 56..111 (folded into tbp)
      int kr = (k < 112) ? krow : 0;
      float v = w1[kr * 64 + bn];
      if (k >= 112) v = 0.f;
      bvals[q] = (f16)v;
    }
    __syncthreads();
    *(v4h*)(As + ar * LDSTR + ac0) = pa;
    v8h pb0, pb1;
#pragma unroll
    for (int q = 0; q < 8; ++q) { pb0[q] = bvals[q]; pb1[q] = bvals[8 + q]; }
    *(v8h*)(BsT + bn * LDSTR + bk0) = pb0;
    *(v8h*)(BsT + bn * LDSTR + bk0 + 8) = pb1;
    __syncthreads();
    v16h a = frag_a(As, 0);
    v16h b = frag_bT(BsT, wave * 16);
    c = WMMA_F16(a, b, c);
  }
  const int lane = tid & 31;
  const int rbase = (lane >> 4) * 8;
  const int n = wave * 16 + (lane & 15);
  const float b1n = b1[n];  // hoisted (invariant over r)
#pragma unroll
  for (int r = 0; r < 8; ++r) {
    int row = rbase + r;
    int m = m0 + row;
    int mm = m < Mrows ? m : Mrows - 1;  // clamped, branchless
    int b = mm / Lh;
    float v = c[r] + b1n + tbp[(size_t)b * 64 + n];
    v = v > 0.f ? v : 0.f;
    hid[row][n] = v;  // garbage rows never consumed (logits store is guarded)
  }
  __syncthreads();
  if (tid < 16) {
    int m = m0 + tid;
    if (m < Mrows) {
      float acc = b2[0];
#pragma unroll
      for (int j = 0; j < 64; ++j) acc += hid[tid][j] * w2[j];
      logits[m] = acc;
    }
  }
}

// ---------------- DIN masked softmax + weighted sum ----------------
__global__ void din_softmax_sum(const float* __restrict__ logits,
                                const int* __restrict__ ids, int padid,
                                const float* __restrict__ val,  // [B,L,D]
                                float* __restrict__ out) {      // [B,D]
  int b = blockIdx.x, tid = threadIdx.x;
  __shared__ float w[Lh];
  if (tid < Lh) {
    float lg = logits[(size_t)b * Lh + tid];
    if (ids[(size_t)b * Lh + tid] == padid) lg = -10000.f;
    w[tid] = lg;
  }
  __syncthreads();
  if (tid == 0) {
    float mx = -1e30f;
    for (int l = 0; l < Lh; ++l) mx = fmaxf(mx, w[l]);
    float s = 0.f;
    for (int l = 0; l < Lh; ++l) { float e = __expf(w[l] - mx); w[l] = e; s += e; }
    float inv = 1.f / s;
    for (int l = 0; l < Lh; ++l) w[l] *= inv;
  }
  __syncthreads();
  if (tid < Dm) {
    float acc = 0.f;
    for (int l = 0; l < Lh; ++l) acc += w[l] * val[((size_t)b * Lh + l) * Dm + tid];
    out[(size_t)b * Dm + tid] = acc;
  }
}

// ---------------- Fused per-sample causal self-attention ----------------
__global__ void attn_kernel(const float* __restrict__ he, const int* __restrict__ hist,
                            const float* __restrict__ wq, const float* __restrict__ wk,
                            const float* __restrict__ wv, float* __restrict__ att) {
  const int b = blockIdx.x, tid = threadIdx.x;
  __shared__ float sh[Lh][Dm];
  __shared__ float sq[Lh][Dm], sk[Lh][Dm], sv[Lh][Dm];
  __shared__ float swq[Dm][Dm], swk[Dm][Dm], swv[Dm][Dm];
  __shared__ float sp[Lh][Lh + 1];
  __shared__ int sid[Lh];
  for (int i = tid; i < Lh * Dm; i += 256) sh[i / Dm][i % Dm] = he[(size_t)b * Lh * Dm + i];
  for (int i = tid; i < Dm * Dm; i += 256) {
    swq[i / Dm][i % Dm] = wq[i]; swk[i / Dm][i % Dm] = wk[i]; swv[i / Dm][i % Dm] = wv[i];
  }
  if (tid < Lh) sid[tid] = hist[(size_t)b * Lh + tid];
  __syncthreads();
  for (int i = tid; i < 3 * Lh * Dm; i += 256) {
    int which = i / (Lh * Dm), rem = i - which * Lh * Dm;
    int r = rem / Dm, d = rem % Dm;
    const float (*W)[Dm] = which == 0 ? swq : which == 1 ? swk : swv;
    float acc = 0.f;
    for (int k = 0; k < Dm; ++k) acc += sh[r][k] * W[k][d];
    (which == 0 ? sq : which == 1 ? sk : sv)[r][d] = acc;
  }
  __syncthreads();
  const float scale = 0.1889822365046136f;  // 1/sqrt(28)
  for (int i = tid; i < Lh * Lh; i += 256) {
    int r = i / Lh, j = i % Lh;
    float acc = 0.f;
    for (int k = 0; k < Dm; ++k) acc += sq[r][k] * sk[j][k];
    acc *= scale;
    if (j > r) acc += -10000.f;
    if (sid[j] == PAD_IDX) acc = -10000.f;
    sp[r][j] = acc;
  }
  __syncthreads();
  if (tid < Lh) {
    float mx = -1e30f;
    for (int j = 0; j < Lh; ++j) mx = fmaxf(mx, sp[tid][j]);
    float s = 0.f;
    for (int j = 0; j < Lh; ++j) { float e = __expf(sp[tid][j] - mx); sp[tid][j] = e; s += e; }
    float inv = 1.f / s;
    for (int j = 0; j < Lh; ++j) sp[tid][j] *= inv;
  }
  __syncthreads();
  for (int i = tid; i < Lh * Dm; i += 256) {
    int r = i / Dm, d = i % Dm;
    float acc = 0.f;
    for (int j = 0; j < Lh; ++j) acc += sp[r][j] * sv[j][d];
    att[((size_t)b * Lh + r) * Dm + d] = acc;
  }
}

// ---------------- small elementwise / gather kernels ----------------
__global__ void gather_seq(const int* __restrict__ hist, const int* __restrict__ ihist,
                           const float* __restrict__ hist_emb, const float* __restrict__ rater_emb,
                           float* __restrict__ he, float* __restrict__ re, int total) {
  int idx = blockIdx.x * blockDim.x + threadIdx.x;
  if (idx >= total) return;
  int d = idx % Dm, bl = idx / Dm;
  he[idx] = hist_emb[(size_t)hist[bl] * Dm + d];
  re[idx] = rater_emb[(size_t)ihist[bl] * Dm + d];
}
__global__ void gather_ui(const int* __restrict__ uid, const int* __restrict__ mid,
                          const float* __restrict__ uet, const float* __restrict__ iet,
                          float* __restrict__ ue, float* __restrict__ ie, int total) {
  int idx = blockIdx.x * blockDim.x + threadIdx.x;
  if (idx >= total) return;
  int d = idx % Dm, b = idx / Dm;
  ue[idx] = uet[(size_t)uid[b] * Dm + d];
  ie[idx] = iet[(size_t)mid[b] * Dm + d];
}
__global__ void tbw_prep(const float* __restrict__ w1u, const float* __restrict__ w1i,
                         float* __restrict__ tbu, float* __restrict__ tbi) {
  int idx = blockIdx.x * blockDim.x + threadIdx.x;
  if (idx >= 28 * 64) return;
  int j = idx / 64, n = idx % 64;
  tbu[idx] = w1u[(56 + j) * 64 + n] + w1u[(84 + j) * 64 + n];
  tbi[idx] = w1i[(56 + j) * 64 + n] + w1i[(84 + j) * 64 + n];
}
__global__ void gfield_k(const float* __restrict__ gate, const float* __restrict__ ge,
                         const float* __restrict__ ie, float* __restrict__ gf, int total) {
  int idx = blockIdx.x * blockDim.x + threadIdx.x;
  if (idx >= total) return;
  float g = gate[idx];
  gf[idx] = g * ge[idx] + (1.f - g) * ie[idx];
}
__global__ void concat_k(float* __restrict__ out, int chunks, int Btot,
                         const float* s0, const float* s1, const float* s2, const float* s3,
                         const float* s4, const float* s5, const float* s6) {
  int idx = blockIdx.x * blockDim.x + threadIdx.x;
  int total = Btot * chunks * Dm;
  if (idx >= total) return;
  int d = idx % Dm;
  int rem = idx / Dm;
  int c = rem % chunks;
  int b = rem / chunks;
  const float* s = c == 0 ? s0 : c == 1 ? s1 : c == 2 ? s2 : c == 3 ? s3 : c == 4 ? s4 : c == 5 ? s5 : s6;
  out[idx] = s[(size_t)b * Dm + d];
}
__global__ void cross_update(const float* __restrict__ x0, const float* __restrict__ xw,
                             const float* __restrict__ g, float* __restrict__ x, int total) {
  int idx = blockIdx.x * blockDim.x + threadIdx.x;
  if (idx >= total) return;
  float gv = g[idx];
  x[idx] = gv * (x0[idx] * xw[idx]) + (1.f - gv) * x[idx];
}
__global__ void combined_k(const float* __restrict__ x, const float* __restrict__ us,
                           const float* __restrict__ its, float* __restrict__ out, int Btot) {
  int idx = blockIdx.x * blockDim.x + threadIdx.x;
  int total = Btot * 388;
  if (idx >= total) return;
  int j = idx % 388, b = idx / 388;
  float v;
  if (j < 196)      v = x[(size_t)b * 196 + j];
  else if (j < 260) v = us[(size_t)b * 64 + (j - 196)];
  else if (j < 324) v = its[(size_t)b * 64 + (j - 260)];
  else { int jj = j - 324; v = us[(size_t)b * 64 + jj] * its[(size_t)b * 64 + jj]; }
  out[idx] = v;
}
__global__ void final_k(const float* __restrict__ t3, const float* __restrict__ w4,
                        const float* __restrict__ b4, float* __restrict__ out, int Btot) {
  int b = blockIdx.x * blockDim.x + threadIdx.x;
  if (b >= Btot) return;
  float acc = b4[0];
#pragma unroll
  for (int j = 0; j < 64; ++j) acc += t3[(size_t)b * 64 + j] * w4[j];
  out[b] = acc;
}

// ---------------- param index map (_make_params insertion order, d_in[10..72]) ----------------
enum {
  P_USER_EMBED = 0, P_ITEM_EMBED, P_HIST_EMBED, P_RATER_EMBED,
  P_RATING_W, P_RATING_B, P_HQ, P_HK, P_HV,
  P_DIN_W1, P_DIN_B1, P_DIN_W2, P_DIN_B2,
  P_IDIN_W1, P_IDIN_B1, P_IDIN_W2, P_IDIN_B2,
  P_BM_W1, P_BM_B1, P_BM_W2, P_BM_B2,
  P_GENRE_W, P_GENRE_B,
  P_GN_W1, P_GN_B1, P_GN_W2, P_GN_B2, P_GN_W3, P_GN_B3,
  P_GG_W, P_GG_B,
  P_CW1, P_CB1, P_CG1W, P_CG1B,
  P_CW2, P_CB2, P_CG2W, P_CG2B,
  P_CW3, P_CB3, P_CG3W, P_CG3B,
  P_CW4, P_CB4, P_CG4W, P_CG4B,
  P_US_W1, P_US_B1, P_US_W2, P_US_B2,
  P_IS_W1, P_IS_B1, P_IS_W2, P_IS_B2,
  P_TM_W1, P_TM_B1, P_TM_W2, P_TM_B2,
  P_TM_W3, P_TM_B3, P_TM_W4, P_TM_B4,
  P_COUNT
};

static inline int cdiv(long a, long b) { return (int)((a + b - 1) / b); }

static void gemm(hipStream_t s, const float* A, int lda, const float* Bw, int ldb,
                 const float* bias, float* C, int ldc, int M, int N, int K, int act) {
  dim3 g(cdiv(M, 32), cdiv(N, 64));
  gemm_bias_act<<<g, dim3(256), 0, s>>>(A, lda, Bw, ldb, bias, C, ldc, M, N, K, act);
}

extern "C" void kernel_launch(void* const* d_in, const int* in_sizes, int n_in,
                              void* d_out, int out_size, void* d_ws, size_t ws_size,
                              hipStream_t stream) {
  (void)n_in; (void)out_size; (void)ws_size;
  const int Btot = in_sizes[0];
  const size_t BL = (size_t)Btot * Lh;

  const int*   user_id   = (const int*)d_in[0];
  const int*   movie_id  = (const int*)d_in[1];
  const float* dense     = (const float*)d_in[2];
  const int*   history   = (const int*)d_in[3];
  const float* hist_rat  = (const float*)d_in[4];
  const float* genres    = (const float*)d_in[5];
  const int*   item_hist = (const int*)d_in[6];
  const float* ih_rat    = (const float*)d_in[7];
  const float* genome    = (const float*)d_in[8];
  // d_in[9] = has_genome_mask (unused by reference)
  const float* P[P_COUNT];
  for (int i = 0; i < P_COUNT; ++i) P[i] = (const float*)d_in[10 + i];

  // ---- workspace carve (all f32) ----
  float* cur = (float*)d_ws;
  auto alloc = [&](size_t n) { float* p = cur; cur += n; return p; };
  float* he   = alloc(BL * Dm);
  float* re   = alloc(BL * Dm);
  float* att  = alloc(BL * Dm);
  float* ue   = alloc((size_t)Btot * Dm);
  float* ie   = alloc((size_t)Btot * Dm);
  float* tbwu = alloc(28 * 64);
  float* tbwi = alloc(28 * 64);
  float* tbpu = alloc((size_t)Btot * 64);
  float* tbpi = alloc((size_t)Btot * 64);
  float* lgu  = alloc(BL);
  float* lgi  = alloc(BL);
  float* uhe  = alloc((size_t)Btot * Dm);
  float* ihe  = alloc((size_t)Btot * Dm);
  float* dh1  = alloc((size_t)Btot * 128);
  float* de   = alloc((size_t)Btot * Dm);
  float* gre  = alloc((size_t)Btot * Dm);
  float* g1   = alloc((size_t)Btot * 256);
  float* g2   = alloc((size_t)Btot * 64);
  float* gne  = alloc((size_t)Btot * Dm);
  float* gate = alloc((size_t)Btot * Dm);
  float* gf   = alloc((size_t)Btot * Dm);
  float* x0   = alloc((size_t)Btot * CDm);
  float* x    = alloc((size_t)Btot * CDm);
  float* xw   = alloc((size_t)Btot * CDm);
  float* gl   = alloc((size_t)Btot * CDm);
  float* usin = alloc((size_t)Btot * 84);
  float* ush  = alloc((size_t)Btot * 256);
  float* us   = alloc((size_t)Btot * 64);
  float* isin = alloc((size_t)Btot * 112);
  float* ish  = alloc((size_t)Btot * 256);
  float* iss  = alloc((size_t)Btot * 64);
  float* comb = alloc((size_t)Btot * 388);
  float* t1   = alloc((size_t)Btot * 256);
  float* t2   = alloc((size_t)Btot * 128);
  float* t3   = alloc((size_t)Btot * 64);

  // ---- embeddings ----
  gather_seq<<<cdiv(BL * Dm, 256), 256, 0, stream>>>(history, item_hist,
      P[P_HIST_EMBED], P[P_RATER_EMBED], he, re, (int)(BL * Dm));
  gather_ui<<<cdiv((size_t)Btot * Dm, 256), 256, 0, stream>>>(user_id, movie_id,
      P[P_USER_EMBED], P[P_ITEM_EMBED], ue, ie, Btot * Dm);

  // ---- causal self-attention over history ----
  attn_kernel<<<Btot, 256, 0, stream>>>(he, history, P[P_HQ], P[P_HK], P[P_HV], att);

  // ---- DIN (user + item side) ----
  tbw_prep<<<cdiv(28 * 64, 256), 256, 0, stream>>>(P[P_DIN_W1], P[P_IDIN_W1], tbwu, tbwi);
  gemm(stream, ie, Dm, tbwu, 64, nullptr, tbpu, 64, Btot, 64, Dm, 0);
  gemm(stream, ue, Dm, tbwi, 64, nullptr, tbpi, 64, Btot, 64, Dm, 0);
  din_logits<<<cdiv(BL, 16), 128, 0, stream>>>(att, hist_rat, P[P_RATING_W], P[P_RATING_B],
      ie, P[P_DIN_W1], P[P_DIN_B1], P[P_DIN_W2], P[P_DIN_B2], tbpu, lgu, (int)BL);
  din_logits<<<cdiv(BL, 16), 128, 0, stream>>>(re, ih_rat, P[P_RATING_W], P[P_RATING_B],
      ue, P[P_IDIN_W1], P[P_IDIN_B1], P[P_IDIN_W2], P[P_IDIN_B2], tbpi, lgi, (int)BL);
  din_softmax_sum<<<Btot, 64, 0, stream>>>(lgu, history, PAD_IDX, att, uhe);
  din_softmax_sum<<<Btot, 64, 0, stream>>>(lgi, item_hist, UPAD_IDX, re, ihe);

  // ---- side features ----
  gemm(stream, dense, 17, P[P_BM_W1], 128, P[P_BM_B1], dh1, 128, Btot, 128, 17, 1);
  gemm(stream, dh1, 128, P[P_BM_W2], Dm, P[P_BM_B2], de, Dm, Btot, Dm, 128, 1);
  gemm(stream, genres, 20, P[P_GENRE_W], Dm, P[P_GENRE_B], gre, Dm, Btot, Dm, 20, 1);
  gemm(stream, genome, 1128, P[P_GN_W1], 256, P[P_GN_B1], g1, 256, Btot, 256, 1128, 1);
  gemm(stream, g1, 256, P[P_GN_W2], 64, P[P_GN_B2], g2, 64, Btot, 64, 256, 1);
  gemm(stream, g2, 64, P[P_GN_W3], Dm, P[P_GN_B3], gne, Dm, Btot, Dm, 64, 0);
  gemm(stream, gne, Dm, P[P_GG_W], Dm, P[P_GG_B], gate, Dm, Btot, Dm, Dm, 2);
  gfield_k<<<cdiv((size_t)Btot * Dm, 256), 256, 0, stream>>>(gate, gne, ie, gf, Btot * Dm);

  // ---- DCN-v2 gated cross layers ----
  concat_k<<<cdiv((size_t)Btot * CDm, 256), 256, 0, stream>>>(x0, 7, Btot,
      ue, ie, uhe, ihe, de, gre, gf);
  hipMemcpyAsync(x, x0, (size_t)Btot * CDm * sizeof(float), hipMemcpyDeviceToDevice, stream);
  const int cwI[4] = {P_CW1, P_CW2, P_CW3, P_CW4};
  const int cbI[4] = {P_CB1, P_CB2, P_CB3, P_CB4};
  const int cgwI[4] = {P_CG1W, P_CG2W, P_CG3W, P_CG4W};
  const int cgbI[4] = {P_CG1B, P_CG2B, P_CG3B, P_CG4B};
  for (int i = 0; i < 4; ++i) {
    gemm(stream, x, CDm, P[cwI[i]], CDm, P[cbI[i]], xw, CDm, Btot, CDm, CDm, 0);
    gemm(stream, x, CDm, P[cgwI[i]], CDm, P[cgbI[i]], gl, CDm, Btot, CDm, CDm, 2);
    cross_update<<<cdiv((size_t)Btot * CDm, 256), 256, 0, stream>>>(x0, xw, gl, x, Btot * CDm);
  }

  // ---- streams ----
  concat_k<<<cdiv((size_t)Btot * 84, 256), 256, 0, stream>>>(usin, 3, Btot,
      ue, uhe, de, nullptr, nullptr, nullptr, nullptr);
  gemm(stream, usin, 84, P[P_US_W1], 256, P[P_US_B1], ush, 256, Btot, 256, 84, 1);
  gemm(stream, ush, 256, P[P_US_W2], 64, P[P_US_B2], us, 64, Btot, 64, 256, 1);
  concat_k<<<cdiv((size_t)Btot * 112, 256), 256, 0, stream>>>(isin, 4, Btot,
      ie, ihe, gre, gf, nullptr, nullptr, nullptr);
  gemm(stream, isin, 112, P[P_IS_W1], 256, P[P_IS_B1], ish, 256, Btot, 256, 112, 1);
  gemm(stream, ish, 256, P[P_IS_W2], 64, P[P_IS_B2], iss, 64, Btot, 64, 256, 1);

  // ---- top MLP ----
  combined_k<<<cdiv((size_t)Btot * 388, 256), 256, 0, stream>>>(x, us, iss, comb, Btot);
  gemm(stream, comb, 388, P[P_TM_W1], 256, P[P_TM_B1], t1, 256, Btot, 256, 388, 1);
  gemm(stream, t1, 256, P[P_TM_W2], 128, P[P_TM_B2], t2, 128, Btot, 128, 256, 1);
  gemm(stream, t2, 128, P[P_TM_W3], 64, P[P_TM_B3], t3, 64, Btot, 64, 128, 1);
  final_k<<<cdiv(Btot, 256), 256, 0, stream>>>(t3, P[P_TM_W4], P[P_TM_B4], (float*)d_out, Btot);
}
